// ConditionalFeedForward_70901320122868
// MI455X (gfx1250) — matrix operand order repaired and checked by reading, not compiled, not run on hardware
//
#include <hip/hip_runtime.h>
#include <hip/hip_bf16.h>
#include <cstdint>

// ---------------- problem constants ----------------
constexpr int E      = 8;
constexpr int N_TOK  = 4096;
constexpr int DIM    = 2048;
constexpr int FFN    = 5632;
constexpr int MAXROW = N_TOK * 2;          // TOP_K = 2 -> exactly 8192 routed rows

typedef __attribute__((ext_vector_type(16))) _Float16 v16h;
typedef __attribute__((ext_vector_type(8)))  _Float16 v8h;
typedef __attribute__((ext_vector_type(4)))  _Float16 v4h;
typedef __attribute__((ext_vector_type(8)))  float    v8f;

// ---------------- routing / compaction ----------------
__global__ void k_init(int* counts, int* counts2) {
    int i = threadIdx.x;
    if (i < E) { counts[i] = 0; counts2[i] = 0; }
}

__global__ void k_route_count(const int* __restrict__ map, int* __restrict__ counts) {
    int t = blockIdx.x * blockDim.x + threadIdx.x;
    if (t >= N_TOK) return;
    #pragma unroll
    for (int e = 0; e < E; ++e)
        if (map[t * E + e]) atomicAdd(&counts[e], 1);
}

__global__ void k_scan(const int* __restrict__ counts, int* __restrict__ offsets) {
    if (threadIdx.x == 0) {
        int acc = 0;
        #pragma unroll
        for (int e = 0; e < E; ++e) { offsets[e] = acc; acc += counts[e]; }
        offsets[E] = acc;
    }
}

__global__ void k_assign(const int* __restrict__ map, const int* __restrict__ offsets,
                         int* __restrict__ counts2, int* __restrict__ tok_of_row,
                         int* __restrict__ row_of) {
    int t = blockIdx.x * blockDim.x + threadIdx.x;
    if (t >= N_TOK) return;
    #pragma unroll
    for (int e = 0; e < E; ++e) {
        if (map[t * E + e]) {
            int pos = atomicAdd(&counts2[e], 1);
            int r = offsets[e] + pos;
            tok_of_row[r] = t;
            row_of[t * E + e] = r;
        }
    }
}

// ---------------- x : f32 -> f16 ----------------
__global__ void k_cvt_x(const float* __restrict__ x, _Float16* __restrict__ xh) {
    int i = blockIdx.x * blockDim.x + threadIdx.x;
    int idx = i * 4;
    if (idx >= N_TOK * DIM) return;
    float4 v = *(const float4*)(x + idx);
    v4h o;
    o[0] = (_Float16)v.x; o[1] = (_Float16)v.y;
    o[2] = (_Float16)v.z; o[3] = (_Float16)v.w;
    *(v4h*)(xh + idx) = o;
}

// ---------------- fragment helpers ----------------
// A fragment (16x32 f16, ISA layout): lane L supplies row m=L%16,
// k = (j/8)*16 + (L/16)*8 + (j%8)  -> two contiguous 8-half (16B) runs.
__device__ inline v16h load_a_frag(const _Float16* __restrict__ p) {
    v8h lo = *(const v8h*)p;
    v8h hi = *(const v8h*)(p + 16);
    v16h a;
    #pragma unroll
    for (int j = 0; j < 8; ++j) { a[j] = lo[j]; a[j + 8] = hi[j]; }
    return a;
}

// Cooperative staging of a 64(n) x 64(k) f32 weight tile into LDS as f16,
// stored in WMMA B-fragment order:
//   half index = ((s*4 + kh)*16 + n16)*16 + j   (s = n/16, kh = k/16, j = k%16)
// Each of the 256 threads: one row x 16-k segment = 4x16B global loads
// (one s_clause), 8 v_cvt_pk, 2x16B ds_store.
__device__ inline void stage_w64(const float* __restrict__ g, int stride,
                                 _Float16* __restrict__ lds, int tid) {
    int n   = tid >> 2;           // 0..63
    int k16 = (tid & 3) * 16;     // 0,16,32,48
    const float4* q = (const float4*)(g + (size_t)n * stride + k16);
    float4 a = q[0], b = q[1], c = q[2], d = q[3];
    v8h lo, hi;
    lo[0] = (_Float16)a.x; lo[1] = (_Float16)a.y;
    lo[2] = (_Float16)a.z; lo[3] = (_Float16)a.w;
    lo[4] = (_Float16)b.x; lo[5] = (_Float16)b.y;
    lo[6] = (_Float16)b.z; lo[7] = (_Float16)b.w;
    hi[0] = (_Float16)c.x; hi[1] = (_Float16)c.y;
    hi[2] = (_Float16)c.z; hi[3] = (_Float16)c.w;
    hi[4] = (_Float16)d.x; hi[5] = (_Float16)d.y;
    hi[6] = (_Float16)d.z; hi[7] = (_Float16)d.w;
    int s = n >> 4, n16 = n & 15, kh = k16 >> 4;
    _Float16* p = lds + (((s * 4 + kh) * 16 + n16) * 16);
    *(v8h*)p       = lo;
    *(v8h*)(p + 8) = hi;
}

// Read one 16-half B fragment (two ds_load_b128: k-blocks kc16 and kc16+1, j offset 8*(lane/16)).
// kc16 = (k_chunk_base/16) within the 64-wide tile: 0 for k 0..31, 2 for k 32..63.
__device__ inline v16h read_b_frag(const _Float16* __restrict__ lds, int s, int kc16, int lane) {
    int n16 = lane & 15, kh = kc16;
    const _Float16* p = lds + (((s * 4 + kh) * 16 + n16) * 16) + ((lane >> 4) * 8);
    v8h lo = *(const v8h*)p;
    v8h hi = *(const v8h*)(p + 256);
    v16h b;
    #pragma unroll
    for (int j = 0; j < 8; ++j) { b[j] = lo[j]; b[j + 8] = hi[j]; }
    return b;
}

__device__ inline v8f wmma_f16(v16h a, v16h b, v8f c) {
    return __builtin_amdgcn_wmma_f32_16x16x32_f16(false, a, false, b, (short)0, c, false, false);
}

// ---------------- GEMM 1+3 fused: h = silu(x@w1^T) * (x@w3^T) ----------------
// block: 256 thr = 8 waves, 4(M)x2(N) -> tile M=64, N=64 (per matrix), K-step 64,
// manually 2x-unrolled K loop (128 per trip) to kill the A-register rotation.
constexpr int BM = 64;
constexpr int BN = 64;

__global__ __launch_bounds__(256) void k_gemm13(
    const _Float16* __restrict__ xh,
    const float* __restrict__ w1, const float* __restrict__ w3,
    const int* __restrict__ offsets, const int* __restrict__ tok_of_row,
    _Float16* __restrict__ hbuf)
{
    const int e        = blockIdx.z;
    const int rowstart = offsets[e];
    const int cnt      = offsets[e + 1] - rowstart;
    const int m0       = blockIdx.y * BM;
    if (m0 >= cnt) return;                          // uniform: safe vs barriers
    const int n0   = blockIdx.x * BN;
    const int tid  = threadIdx.x;
    const int lane = tid & 31;
    const int wid  = tid >> 5;
    const int wm   = wid & 3;
    const int wn   = wid >> 2;

    __shared__ _Float16 ls[2][2][4096];             // [buf][w1/w3][64x64 tile] = 32 KB

    // A row for this lane
    int mrow = m0 + wm * 16 + (lane & 15);
    int mcl  = mrow < cnt ? mrow : cnt - 1;
    const int tok = tok_of_row[rowstart + mcl];
    const _Float16* arow = xh + (size_t)tok * DIM + ((lane >> 4) * 8);

    const size_t wbase = (size_t)e * FFN * DIM + (size_t)n0 * DIM;
    const float* g1 = w1 + wbase;
    const float* g3 = w3 + wbase;

    const int sA = wn * 2, sB = sA + 1;
    v8f a1a = {}, a1b = {}, a3a = {}, a3b = {};
    v16h xA0, xA1, yA0, yA1;                        // ping-pong A register sets

    // prologue
    xA0 = load_a_frag(arow);
    xA1 = load_a_frag(arow + 32);
    stage_w64(g1, DIM, &ls[0][0][0], tid);
    stage_w64(g3, DIM, &ls[0][1][0], tid);
    __syncthreads();

    // one sub-step: consume buffer BUF at k-base KB with A regs (A0,A1);
    // prefetch KB+64 into the other buffer and (NA0,NA1).
    auto gstep = [&](int BUF, int kb, v16h& A0, v16h& A1, v16h& NA0, v16h& NA1) {
        const int nk = kb + 64;
        if (nk < DIM) {
            stage_w64(g1 + nk, DIM, &ls[BUF ^ 1][0][0], tid);
            stage_w64(g3 + nk, DIM, &ls[BUF ^ 1][1][0], tid);
            NA0 = load_a_frag(arow + nk);
            NA1 = load_a_frag(arow + nk + 32);
        }
        // batch all 8 B-fragment LDS reads, then 8 WMMAs
        v16h b0 = read_b_frag(&ls[BUF][0][0], sA, 0, lane);
        v16h b1 = read_b_frag(&ls[BUF][0][0], sB, 0, lane);
        v16h b2 = read_b_frag(&ls[BUF][1][0], sA, 0, lane);
        v16h b3 = read_b_frag(&ls[BUF][1][0], sB, 0, lane);
        v16h b4 = read_b_frag(&ls[BUF][0][0], sA, 2, lane);
        v16h b5 = read_b_frag(&ls[BUF][0][0], sB, 2, lane);
        v16h b6 = read_b_frag(&ls[BUF][1][0], sA, 2, lane);
        v16h b7 = read_b_frag(&ls[BUF][1][0], sB, 2, lane);
        a1a = wmma_f16(A0, b0, a1a);
        a1b = wmma_f16(A0, b1, a1b);
        a3a = wmma_f16(A0, b2, a3a);
        a3b = wmma_f16(A0, b3, a3b);
        a1a = wmma_f16(A1, b4, a1a);
        a1b = wmma_f16(A1, b5, a1b);
        a3a = wmma_f16(A1, b6, a3a);
        a3b = wmma_f16(A1, b7, a3b);
        __syncthreads();
    };

    #pragma unroll 1
    for (int k0 = 0; k0 < DIM; k0 += 128) {         // DIM/128 = 16 exact
        gstep(0, k0,      xA0, xA1, yA0, yA1);
        gstep(1, k0 + 64, yA0, yA1, xA0, xA1);
    }

    // epilogue: SwiGLU, store f16.  C layout: vgpr v -> m = v + 8*(lane/16), n = lane%16
    const int nA = n0 + wn * 32 + (lane & 15);
    const int nB = nA + 16;
    const int mbase = m0 + wm * 16 + 8 * (lane >> 4);
    #pragma unroll
    for (int v = 0; v < 8; ++v) {
        int mloc = mbase + v;
        if (mloc < cnt) {
            size_t r = (size_t)(rowstart + mloc) * FFN;
            float g = a1a[v];
            float hv = (g / (1.0f + __expf(-g))) * a3a[v];
            hbuf[r + nA] = (_Float16)hv;
            g = a1b[v];
            hv = (g / (1.0f + __expf(-g))) * a3b[v];
            hbuf[r + nB] = (_Float16)hv;
        }
    }
}

// ---------------- GEMM 2: y = gate * (h @ w2^T) ----------------
__global__ __launch_bounds__(256) void k_gemm2(
    const _Float16* __restrict__ hbuf, const float* __restrict__ w2,
    const float* __restrict__ probs,
    const int* __restrict__ offsets, const int* __restrict__ tok_of_row,
    float* __restrict__ y)
{
    const int e        = blockIdx.z;
    const int rowstart = offsets[e];
    const int cnt      = offsets[e + 1] - rowstart;
    const int m0       = blockIdx.y * BM;
    if (m0 >= cnt) return;
    const int n0   = blockIdx.x * BN;
    const int tid  = threadIdx.x;
    const int lane = tid & 31;
    const int wid  = tid >> 5;
    const int wm   = wid & 3;
    const int wn   = wid >> 2;

    __shared__ _Float16 ls[2][4096];                // [buf][64x64 tile] = 16 KB

    int mrow = m0 + wm * 16 + (lane & 15);
    int mcl  = mrow < cnt ? mrow : cnt - 1;
    const _Float16* arow = hbuf + (size_t)(rowstart + mcl) * FFN + ((lane >> 4) * 8);

    const float* g2 = w2 + (size_t)e * DIM * FFN + (size_t)n0 * FFN;

    const int sA = wn * 2, sB = sA + 1;
    v8f aca = {}, acb = {};
    v16h xA0, xA1, yA0, yA1;

    xA0 = load_a_frag(arow);
    xA1 = load_a_frag(arow + 32);
    stage_w64(g2, FFN, &ls[0][0], tid);
    __syncthreads();

    auto gstep = [&](int BUF, int kb, v16h& A0, v16h& A1, v16h& NA0, v16h& NA1) {
        const int nk = kb + 64;
        if (nk < FFN) {
            stage_w64(g2 + nk, FFN, &ls[BUF ^ 1][0], tid);
            NA0 = load_a_frag(arow + nk);
            NA1 = load_a_frag(arow + nk + 32);
        }
        v16h b0 = read_b_frag(&ls[BUF][0], sA, 0, lane);
        v16h b1 = read_b_frag(&ls[BUF][0], sB, 0, lane);
        v16h b2 = read_b_frag(&ls[BUF][0], sA, 2, lane);
        v16h b3 = read_b_frag(&ls[BUF][0], sB, 2, lane);
        aca = wmma_f16(A0, b0, aca);
        acb = wmma_f16(A0, b1, acb);
        aca = wmma_f16(A1, b2, aca);
        acb = wmma_f16(A1, b3, acb);
        __syncthreads();
    };

    #pragma unroll 1
    for (int k0 = 0; k0 < FFN; k0 += 128) {         // FFN/128 = 44 exact
        gstep(0, k0,      xA0, xA1, yA0, yA1);
        gstep(1, k0 + 64, yA0, yA1, xA0, xA1);
    }

    const int nA = n0 + wn * 32 + (lane & 15);
    const int nB = nA + 16;
    const int mbase = m0 + wm * 16 + 8 * (lane >> 4);
    #pragma unroll
    for (int v = 0; v < 8; ++v) {
        int mloc = mbase + v;
        if (mloc < cnt) {
            int r = rowstart + mloc;
            int t = tok_of_row[r];
            float gate = probs[t * E + e];
            y[(size_t)r * DIM + nA] = gate * aca[v];
            y[(size_t)r * DIM + nB] = gate * acb[v];
        }
    }
}

// ---------------- combine: out[t] = sum over routed experts of y[row] ----------------
__global__ void k_combine(const int* __restrict__ map, const int* __restrict__ row_of,
                          const float* __restrict__ y, float* __restrict__ out)
{
    int i = blockIdx.x * blockDim.x + threadIdx.x;
    if (i >= N_TOK * DIM) return;
    int t = i / DIM, d = i - t * DIM;
    float s = 0.0f;
    #pragma unroll
    for (int e = 0; e < E; ++e) {
        if (map[t * E + e]) {
            int r = row_of[t * E + e];
            s += y[(size_t)r * DIM + d];
        }
    }
    out[i] = s;
}

// ---------------- launch ----------------
extern "C" void kernel_launch(void* const* d_in, const int* in_sizes, int n_in,
                              void* d_out, int out_size, void* d_ws, size_t ws_size,
                              hipStream_t stream) {
    const float* x     = (const float*)d_in[0];
    const int*   map   = (const int*)  d_in[1];
    const float* probs = (const float*)d_in[2];
    const float* w1    = (const float*)d_in[3];
    const float* w2    = (const float*)d_in[4];
    const float* w3    = (const float*)d_in[5];
    float*       out   = (float*)d_out;

    // workspace carve-up (total ~176.5 MB)
    char* ws = (char*)d_ws;
    size_t cur = 0;
    auto alloc = [&](size_t bytes) -> void* {
        cur = (cur + 255) & ~(size_t)255;
        void* p = ws + cur;
        cur += bytes;
        return p;
    };
    int*       counts     = (int*)alloc(E * sizeof(int));
    int*       counts2    = (int*)alloc(E * sizeof(int));
    int*       offsets    = (int*)alloc((E + 1) * sizeof(int));
    int*       tok_of_row = (int*)alloc(MAXROW * sizeof(int));
    int*       row_of     = (int*)alloc((size_t)N_TOK * E * sizeof(int));
    _Float16*  xh         = (_Float16*)alloc((size_t)N_TOK * DIM * sizeof(_Float16));
    _Float16*  hbuf       = (_Float16*)alloc((size_t)MAXROW * FFN * sizeof(_Float16));
    float*     ybuf       = (float*)alloc((size_t)MAXROW * DIM * sizeof(float));
    (void)ws_size;

    // 1) routing compaction
    k_init<<<1, 32, 0, stream>>>(counts, counts2);
    k_route_count<<<(N_TOK + 255) / 256, 256, 0, stream>>>(map, counts);
    k_scan<<<1, 32, 0, stream>>>(counts, offsets);
    k_assign<<<(N_TOK + 255) / 256, 256, 0, stream>>>(map, offsets, counts2, tok_of_row, row_of);

    // 2) x -> f16
    k_cvt_x<<<(N_TOK * DIM / 4 + 255) / 256, 256, 0, stream>>>(x, xh);

    // 3) fused up-projection + SwiGLU  (worst-case grid, early-exit per expert)
    {
        dim3 grid(FFN / BN, N_TOK / BM, E);
        k_gemm13<<<grid, 256, 0, stream>>>(xh, w1, w3, offsets, tok_of_row, hbuf);
    }

    // 4) gated down-projection
    {
        dim3 grid(DIM / BN, N_TOK / BM, E);
        k_gemm2<<<grid, 256, 0, stream>>>(hbuf, w2, probs, offsets, tok_of_row, ybuf);
    }

    // 5) combine routed expert outputs per token
    k_combine<<<(N_TOK * DIM + 255) / 256, 256, 0, stream>>>(map, row_of, ybuf, out);
}
